// SpeakerEncoderLoss_85873576116723
// MI455X (gfx1250) — matrix operand order repaired and checked by reading, not compiled, not run on hardware
//
#include <hip/hip_runtime.h>

// SpeakerEncoderLoss (GE2E) for MI455X / gfx1250, wave32.
// Pipeline:
//   k_prep  : per-speaker stats in fp32 (sums, centroid, norms, leave-one-out
//             cosine => S_same), convert x -> bf16 (paired u32 stores),
//             centroids -> bf16.
//   k_score : fused bf16 WMMA GEMM (centroids x x^T) + scaled cosine +
//             diagonal substitution + shifted logsumexp over the 512 rows.
//             B fragments register-cached (192 VGPRs, invariant over row
//             chunks); A tile staged through LDS; 1.0 frag-loads per WMMA.
//   k_reduce: deterministic single-block sum of per-column (lse - S_same).
//
// Workspace requirement: ~51.5 MB (x_bf16 50.3MB + centroids + scalars).

#define N_SPK 512
#define M_SMP 64
#define D_DIM 768
#define NM    (N_SPK * M_SMP)      // 32768 columns
#define KCH   (D_DIM / 32)          // 24 k-chunks of 32
#define LDA   776                   // padded LDS row stride (ushorts), breaks bank conflicts
#define EPS   1e-8f

typedef __attribute__((ext_vector_type(16))) __bf16 v16bf;
typedef __attribute__((ext_vector_type(8)))  float  v8f;
typedef __attribute__((ext_vector_type(4)))  unsigned int u32x4;

union FragU { v16bf v; u32x4 q[2]; };

// float -> bf16 (round-to-nearest-even), raw ushort
__device__ __forceinline__ unsigned short f2bf(float f) {
    unsigned int u = __float_as_uint(f);
    u = (u + 0x7FFFu + ((u >> 16) & 1u)) >> 16;
    return (unsigned short)u;
}

// A fragment: lane L holds row M=(L&15); half h=(L>>4) selects
// K subset {8h..8h+7} (elems 0-7) and {16+8h..16+8h+7} (elems 8-15).
__device__ __forceinline__ v16bf frag_a(const unsigned short* row, int kk, int h) {
    FragU f;
    const unsigned short* p = row + kk * 32;
    f.q[0] = *(const u32x4*)(p + h * 8);          // k = kk*32 + 8h   .. +7
    f.q[1] = *(const u32x4*)(p + 16 + h * 8);     // k = kk*32+16+8h  .. +7
    return f.v;
}

// B fragment: lane L holds column N=(L&15); half h selects
// K subset {8h..8h+7} (elems 0-7) and {16+8h..16+8h+7} (elems 8-15).
__device__ __forceinline__ v16bf frag_b(const unsigned short* row, int kk, int h) {
    FragU f;
    const unsigned short* p = row + kk * 32;
    f.q[0] = *(const u32x4*)(p + h * 8);          // k = kk*32 + 8h   .. +7
    f.q[1] = *(const u32x4*)(p + 16 + h * 8);     // k = kk*32+16+8h  .. +7
    return f.v;
}

// ---------------------------------------------------------------------------
// Kernel 1: per-speaker prep. grid = 512 blocks x 256 threads.
// ---------------------------------------------------------------------------
__global__ void k_prep(const float* __restrict__ x,
                       const float* __restrict__ wp, const float* __restrict__ bp,
                       unsigned short* __restrict__ xb,   // [NM, D] bf16
                       unsigned short* __restrict__ cb,   // [N, D] bf16 centroids
                       float* __restrict__ invCN,         // [N]
                       float* __restrict__ invXN,         // [NM]
                       float* __restrict__ ssame)         // [NM]
{
    __shared__ float sums[D_DIM];
    __shared__ float red[256];
    __shared__ float pxx[256], pde[256], pee[256];

    const int n = blockIdx.x;
    const int t = threadIdx.x;
    const float w = *wp, b = *bp;
    const float* xr = x + (size_t)n * M_SMP * D_DIM;

    // ---- sums over the 64 samples (each thread owns 3 d-slots) ----
    float s0 = 0.f, s1 = 0.f, s2 = 0.f;
    for (int m = 0; m < M_SMP; ++m) {
        const float* r = xr + m * D_DIM;
        s0 += r[t]; s1 += r[t + 256]; s2 += r[t + 512];
    }
    sums[t] = s0; sums[t + 256] = s1; sums[t + 512] = s2;
    __syncthreads();

    // ---- centroid, centroid norm, bf16 centroid store ----
    float c2 = 0.f;
    for (int i = t; i < D_DIM; i += 256) {
        float c = sums[i] * (1.0f / 64.0f);
        cb[(size_t)n * D_DIM + i] = f2bf(c);
        c2 += c * c;
    }
    red[t] = c2;
    __syncthreads();
    for (int off = 128; off > 0; off >>= 1) {
        if (t < off) red[t] += red[t + off];
        __syncthreads();
    }
    if (t == 0) invCN[n] = 1.0f / fmaxf(sqrtf(red[0]), EPS);

    // ---- per-sample: ||x||, leave-one-out cosine, paired bf16 stores ----
    const int m = t >> 2, sub = t & 3;            // 4 threads per sample
    const float* r = xr + m * D_DIM;
    unsigned int* xbrow = (unsigned int*)(xb + ((size_t)n * M_SMP + m) * D_DIM);
    float xx = 0.f, de = 0.f, ee = 0.f;
    for (int p = sub; p < D_DIM / 2; p += 4) {    // 96 bf16-pairs per thread
        int d = 2 * p;
        float v0 = r[d], v1 = r[d + 1];
        xbrow[p] = (unsigned int)f2bf(v0) | ((unsigned int)f2bf(v1) << 16);
        float e0 = (sums[d]     - v0) * (1.0f / 63.0f);
        float e1 = (sums[d + 1] - v1) * (1.0f / 63.0f);
        xx += v0 * v0 + v1 * v1;
        de += e0 * v0 + e1 * v1;
        ee += e0 * e0 + e1 * e1;
    }
    pxx[t] = xx; pde[t] = de; pee[t] = ee;
    __syncthreads();
    if (sub == 0) {
        float X  = pxx[t] + pxx[t + 1] + pxx[t + 2] + pxx[t + 3];
        float Dd = pde[t] + pde[t + 1] + pde[t + 2] + pde[t + 3];
        float E  = pee[t] + pee[t + 1] + pee[t + 2] + pee[t + 3];
        float xn = fmaxf(sqrtf(X), EPS);
        float en = fmaxf(sqrtf(E), EPS);
        int col  = n * M_SMP + m;
        invXN[col] = 1.0f / xn;
        ssame[col] = w * (Dd / (en * xn)) + b;
    }
}

// ---------------------------------------------------------------------------
// Kernel 2: fused WMMA GEMM + logsumexp.
// grid = 256 blocks x 256 threads (8 waves); block = 128 columns x 512 rows.
// B fragments live in registers for the whole kernel (x_bf16 streamed from
// HBM exactly once); A rows staged via LDS and shared by all 8 waves.
// ---------------------------------------------------------------------------
__global__ __launch_bounds__(256, 1) void k_score(
                        const unsigned short* __restrict__ xb,
                        const unsigned short* __restrict__ cb,
                        const float* __restrict__ invCN,
                        const float* __restrict__ invXN,
                        const float* __restrict__ ssame,
                        const float* __restrict__ wp, const float* __restrict__ bp,
                        float* __restrict__ colval)
{
    __shared__ __align__(16) unsigned short Atile[16 * LDA];   // ~24.8 KB
    __shared__ float invcn_s[N_SPK];                           // 2 KB

    const int t    = threadIdx.x;
    const int lane = t & 31;
    const int wave = t >> 5;
    const int h    = lane >> 4;
    const int l16  = lane & 15;

    const float w = *wp, b = *bp;
    const float shift = w + b;                  // S <= w*1 + b, so exp(S-shift) <= 1

    for (int i = t; i < N_SPK; i += 256) invcn_s[i] = invCN[i];

    const int col = blockIdx.x * 128 + wave * 16 + l16;
    const float invxn_c = invXN[col];
    const float ssame_c = ssame[col];
    const int   jcol    = col >> 6;             // speaker owning this column

    // ---- register-cache all B fragments for this wave's 16 columns ----
    const unsigned short* brow = xb + (size_t)col * D_DIM;
    v16bf bfrag[KCH];
#pragma unroll
    for (int kk = 0; kk < KCH; ++kk)
        bfrag[kk] = frag_b(brow, kk, h);

    float sumexp = 0.0f;

    for (int rc = 0; rc < N_SPK / 16; ++rc) {   // 32 row chunks of 16 centroids
        __syncthreads();                        // prior chunk done with Atile
        // ---- stage 16 centroid rows into LDS (6 b128 chunks per thread) ----
        const unsigned short* asrc = cb + (size_t)rc * 16 * D_DIM;
        for (int i = t; i < 16 * (D_DIM / 8); i += 256) {
            int row = i / (D_DIM / 8), off = i % (D_DIM / 8);
            *(u32x4*)(Atile + row * LDA + off * 8) =
                *(const u32x4*)(asrc + (size_t)row * D_DIM + off * 8);
        }
        __syncthreads();

        const unsigned short* arow = Atile + l16 * LDA;
        v8f acc = {};
#pragma unroll
        for (int kk = 0; kk < KCH; ++kk) {
            v16bf af = frag_a(arow, kk, h);     // 2x ds_load_b128 per wmma
            acc = __builtin_amdgcn_wmma_f32_16x16x32_bf16(
                false, af, false, bfrag[kk], (short)0, acc, false, false);
        }

        // C layout: lane (h,l16) elem r -> row = rc*16 + 8h + r, col = l16
#pragma unroll
        for (int r = 0; r < 8; ++r) {
            int row = rc * 16 + h * 8 + r;
            float s = w * (acc[r] * invcn_s[row] * invxn_c) + b;
            if (row == jcol) s = ssame_c;       // diagonal: leave-one-out score
            sumexp += __expf(s - shift);
        }
    }

    // column stats split across lane pair (l16, l16+16)
    sumexp += __shfl_xor(sumexp, 16, 32);
    float lse = shift + __logf(sumexp);
    if (h == 0) colval[col] = lse - ssame_c;    // per-column loss contribution
}

// ---------------------------------------------------------------------------
// Kernel 3: deterministic final reduction.
// ---------------------------------------------------------------------------
__global__ void k_reduce(const float* __restrict__ colval, float* __restrict__ out)
{
    __shared__ float red[256];
    const int t = threadIdx.x;
    float s = 0.f;
    for (int i = t; i < NM; i += 256) s += colval[i];
    red[t] = s;
    __syncthreads();
    for (int off = 128; off > 0; off >>= 1) {
        if (t < off) red[t] += red[t + off];
        __syncthreads();
    }
    if (t == 0) out[0] = red[0];
}

// ---------------------------------------------------------------------------
extern "C" void kernel_launch(void* const* d_in, const int* in_sizes, int n_in,
                              void* d_out, int out_size, void* d_ws, size_t ws_size,
                              hipStream_t stream)
{
    (void)in_sizes; (void)n_in; (void)out_size; (void)ws_size;

    const float* x  = (const float*)d_in[0];   // [NM, D] fp32
    const float* wp = (const float*)d_in[1];   // scalar
    const float* bp = (const float*)d_in[2];   // scalar

    char* ws = (char*)d_ws;
    size_t off = 0;
    unsigned short* xb = (unsigned short*)(ws + off); off += (size_t)NM * D_DIM * 2;     // 50.33 MB
    unsigned short* cb = (unsigned short*)(ws + off); off += (size_t)N_SPK * D_DIM * 2;  // 0.79 MB
    float* invCN  = (float*)(ws + off); off += (size_t)N_SPK * 4;
    float* invXN  = (float*)(ws + off); off += (size_t)NM * 4;
    float* ssame  = (float*)(ws + off); off += (size_t)NM * 4;
    float* colval = (float*)(ws + off); off += (size_t)NM * 4;   // total ~51.5 MB

    k_prep<<<N_SPK, 256, 0, stream>>>(x, wp, bp, xb, cb, invCN, invXN, ssame);
    k_score<<<NM / 128, 256, 0, stream>>>(xb, cb, invCN, invXN, ssame, wp, bp, colval);
    k_reduce<<<1, 256, 0, stream>>>(colval, (float*)d_out);
}
